// STFORMER_64974265254350
// MI455X (gfx1250) — matrix operand order repaired and checked by reading, not compiled, not run on hardware
//
#include <hip/hip_runtime.h>
#include <cstring>

typedef __attribute__((ext_vector_type(16))) __bf16 v16bf;
typedef __attribute__((ext_vector_type(8)))  __bf16 v8bf;
typedef __attribute__((ext_vector_type(2)))  __bf16 v2bf;
typedef __attribute__((ext_vector_type(8)))  float  v8f;
typedef __attribute__((ext_vector_type(2)))  float  v2f;

// ---------------------------------------------------------------------------
// WMMA fragment maps (CDNA5 ISA 7.12.2, wave32):
//  A 16x32 bf16: lane (m=l&15, half=l>>4): elems 0..7 -> k = 8*half + i,
//                elems 8..15 -> k = 16 + 8*half + (i-8)   (two 16B chunks)
//  B 32x16 bf16: lane col n=l&15: same k map as A (two 16B chunks)
//  C 16x16 f32 : row = vgpr + 8*half, col = l&15
// ---------------------------------------------------------------------------

// Generic batched GEMM:  C = act(alpha * (A @ B) * mask + bias + residual)
// batch offset for tensor T: off(z) = (z/innerN)*T.outer + (z%innerN)*T.inner
// A addressed [m*sAm + k]; B addressed [k*sBk + n*sBn] (lets us do K^T gather)
struct G {
  const float* A; long Ao; long Ai; int sAm;
  const float* B; long Bo; long Bi; int sBk; int sBn;
  float*       C; long Co; long Ci; int sCm;
  const float* bias;
  const float* R;  long Ro; long Ri; int sRm;
  const float* Mk; long Mko; int sMm;
  float alpha; int innerN; int M; int N; int K; int act; int batches;
};

#define LDSTRIDE 40   // bf16 elems per row: 80 B, multiple of 16 B, bank-skewed
#define LDTILE   (64 * LDSTRIDE)

struct StageRegs { v2f a[8]; float b0[8]; float b1[8]; };

__global__ __launch_bounds__(128) void gemm_wmma_kernel(G g) {
  __shared__ __bf16 ldsA[2][LDTILE];   // [m][k], double buffered
  __shared__ __bf16 ldsB[2][LDTILE];   // [n][k] (transposed tile), double buffered
  const int z  = blockIdx.z;
  const int zo = z / g.innerN, zi = z % g.innerN;
  const float* Ab = g.A + (long)zo * g.Ao + (long)zi * g.Ai;
  const float* Bb = g.B + (long)zo * g.Bo + (long)zi * g.Bi;
  float*       Cb = g.C + (long)zo * g.Co + (long)zi * g.Ci;
  const int mb = blockIdx.y * 64, nb = blockIdx.x * 64;
  const int lane = threadIdx.x & 31, w = threadIdx.x >> 5;
  const int ml = lane & 15, half = lane >> 4;
  v8f acc[4] = {};

  // phase 1: issue all 24 global loads for one 64x32 A + 32x64 B tile
  auto loads = [&](int kb) {
    StageRegs r;
    int idx = threadIdx.x;
#pragma unroll
    for (int it = 0; it < 8; ++it, idx += 128) {       // A: float2 along k
      int rr = idx >> 4, cp = idx & 15;
      r.a[it] = *(const v2f*)(Ab + (long)(mb + rr) * g.sAm + kb + 2 * cp);
    }
    idx = threadIdx.x;
#pragma unroll
    for (int it = 0; it < 8; ++it, idx += 128) {       // B: two k-rows per pair
      int n = idx & 63, kp = idx >> 6;
      const float* bp = Bb + (long)(kb + 2 * kp) * g.sBk + (long)(nb + n) * g.sBn;
      r.b0[it] = bp[0]; r.b1[it] = bp[g.sBk];
    }
    return r;
  };
  // phase 2: convert to bf16 and store into LDS buffer
  auto stores = [&](const StageRegs& r, __bf16* lA, __bf16* lB) {
    int idx = threadIdx.x;
#pragma unroll
    for (int it = 0; it < 8; ++it, idx += 128) {
      int rr = idx >> 4, cp = idx & 15;
      v2bf p; p[0] = (__bf16)r.a[it][0]; p[1] = (__bf16)r.a[it][1];
      *(v2bf*)&lA[rr * LDSTRIDE + 2 * cp] = p;
    }
    idx = threadIdx.x;
#pragma unroll
    for (int it = 0; it < 8; ++it, idx += 128) {       // transpose -> [n][k]
      int n = idx & 63, kp = idx >> 6;
      v2bf p; p[0] = (__bf16)r.b0[it]; p[1] = (__bf16)r.b1[it];
      *(v2bf*)&lB[n * LDSTRIDE + 2 * kp] = p;
    }
  };
  // 4 WMMAs from an LDS buffer
  auto compute = [&](const __bf16* lA, const __bf16* lB) {
    const int arow = (w * 16 + ml) * LDSTRIDE;
    v8bf a_lo = *(const v8bf*)&lA[arow + 8 * half];
    v8bf a_hi = *(const v8bf*)&lA[arow + 16 + 8 * half];
    v16bf af = __builtin_shufflevector(a_lo, a_hi, 0, 1, 2, 3, 4, 5, 6, 7,
                                       8, 9, 10, 11, 12, 13, 14, 15);
#pragma unroll
    for (int nt = 0; nt < 4; ++nt) {
      const int brow = (nt * 16 + ml) * LDSTRIDE;
      v8bf b_lo = *(const v8bf*)&lB[brow + 8 * half];
      v8bf b_hi = *(const v8bf*)&lB[brow + 16 + 8 * half];
      v16bf bf = __builtin_shufflevector(b_lo, b_hi, 0, 1, 2, 3, 4, 5, 6, 7,
                                         8, 9, 10, 11, 12, 13, 14, 15);
      acc[nt] = __builtin_amdgcn_wmma_f32_16x16x32_bf16(
          false, af, false, bf, (short)0, acc[nt], false, false);
    }
  };

  // software-pipelined double-buffered main loop
  __bf16 *cA = &ldsA[0][0], *cB = &ldsB[0][0];
  __bf16 *nA = &ldsA[1][0], *nB = &ldsB[1][0];
  { StageRegs r0 = loads(0); stores(r0, cA, cB); }
  __syncthreads();
  for (int kb = 0;;) {
    bool more = (kb + 32) < g.K;
    StageRegs r;
    if (more) r = loads(kb + 32);      // next tile's global loads in flight
    compute(cA, cB);                   // ds_load_b128 + 4x WMMA on current tile
    if (more) stores(r, nA, nB);       // cvt + ds_store next tile
    __syncthreads();
    kb += 32;
    if (kb >= g.K) break;
    __bf16* t;
    t = cA; cA = nA; nA = t;
    t = cB; cB = nB; nB = t;
  }

#pragma unroll
  for (int nt = 0; nt < 4; ++nt) {
#pragma unroll
    for (int q = 0; q < 8; ++q) {
      int row = mb + w * 16 + half * 8 + q;
      int col = nb + nt * 16 + ml;
      float v = acc[nt][q] * g.alpha;
      if (g.Mk)   v *= g.Mk[(long)zo * g.Mko + (long)row * g.sMm + col];
      if (g.bias) v += g.bias[col];
      if (g.R)    v += g.R[(long)zo * g.Ro + (long)zi * g.Ri +
                           (long)row * g.sRm + col];
      if (g.act == 1)      v = 0.5f * v * (1.0f + erff(v * 0.70710678118f));
      else if (g.act == 2) v = fmaxf(v, 0.0f);
      Cb[(long)row * g.sCm + col] = v;
    }
  }
}

// ---------------------------------------------------------------------------
// STAR embed reductions over enc_x [16,12,512,512]
// ---------------------------------------------------------------------------
__global__ __launch_bounds__(512) void colsum_kernel(const float* __restrict__ EX,
                                                     float* __restrict__ SO) {
  int bt = blockIdx.x;               // b*12 + t
  int b = bt / 12, t = bt % 12;
  int j = threadIdx.x;               // d
  const float* base = EX + (long)bt * 512 * 512;
  float acc = 0.f;
  for (int o = 0; o < 512; ++o) acc += base[(long)o * 512 + j];
  SO[((long)(b * 512 + j)) * 12 + t] = acc;
}

__global__ __launch_bounds__(256) void rowsum_kernel(const float* __restrict__ EX,
                                                     float* __restrict__ SD) {
  int gw   = (int)((blockIdx.x * 256 + threadIdx.x) >> 5);  // one wave per row
  int lane = threadIdx.x & 31;
  int b = gw / (12 * 512);
  int rem = gw % (12 * 512);
  int t = rem >> 9, o = rem & 511;
  const float* base = EX + (((long)(b * 12 + t)) * 512 + o) * 512;
  float acc = 0.f;
#pragma unroll
  for (int k = 0; k < 16; ++k) acc += base[lane + 32 * k];
#pragma unroll
  for (int m = 16; m > 0; m >>= 1) acc += __shfl_xor(acc, m, 32);
  if (lane == 0) SD[((long)(b * 512 + o)) * 12 + t] = acc;
}

__global__ __launch_bounds__(256) void embed_kernel(
    const float* __restrict__ SO, const float* __restrict__ SD,
    const float* __restrict__ Wo, const float* __restrict__ bo,
    const float* __restrict__ Wd, const float* __restrict__ bd,
    float* __restrict__ X) {
  long row = blockIdx.x;             // b*512 + n
  __shared__ float so_s[12], sd_s[12];
  int e = threadIdx.x;
  if (e < 12)       so_s[e]      = SO[row * 12 + e];
  else if (e < 24)  sd_s[e - 12] = SD[row * 12 + (e - 12)];
  __syncthreads();
  float acc = 512.0f * (bo[e] + bd[e]);
#pragma unroll
  for (int k = 0; k < 12; ++k)
    acc += so_s[k] * Wo[k * 256 + e] + sd_s[k] * Wd[k * 256 + e];
  X[row * 256 + e] = acc;
}

// ---------------------------------------------------------------------------
// LayerNorm over rows of 256
// ---------------------------------------------------------------------------
__global__ __launch_bounds__(256) void layernorm_kernel(
    const float* __restrict__ x, float* __restrict__ y,
    const float* __restrict__ gam, const float* __restrict__ bet) {
  __shared__ float red[256];
  __shared__ float mean_s, rstd_s;
  long row = blockIdx.x;
  int t = threadIdx.x;
  float v = x[row * 256 + t];
  red[t] = v; __syncthreads();
  for (int s = 128; s > 0; s >>= 1) { if (t < s) red[t] += red[t + s]; __syncthreads(); }
  if (t == 0) mean_s = red[0] * (1.0f / 256.0f);
  __syncthreads();
  float d = v - mean_s;
  red[t] = d * d; __syncthreads();
  for (int s = 128; s > 0; s >>= 1) { if (t < s) red[t] += red[t + s]; __syncthreads(); }
  if (t == 0) rstd_s = rsqrtf(red[0] * (1.0f / 256.0f) + 1e-5f);
  __syncthreads();
  y[row * 256 + t] = d * rstd_s * gam[t] + bet[t];
}

// ---------------------------------------------------------------------------
// Softmax over rows of 512 (in-place).  masked: out = (s!=0) ? softmax(s) : 0
// (softmax normalizer over the full row including zeros, matching reference)
// ---------------------------------------------------------------------------
__global__ __launch_bounds__(256) void softmax_kernel(float* __restrict__ Sp,
                                                      int masked) {
  __shared__ float red[256];
  __shared__ float rmax, rsum;
  long row = blockIdx.x;
  float* p = Sp + row * 512;
  int t = threadIdx.x;
  float a = p[t], b = p[t + 256];
  red[t] = fmaxf(a, b); __syncthreads();
  for (int s = 128; s > 0; s >>= 1) { if (t < s) red[t] = fmaxf(red[t], red[t + s]); __syncthreads(); }
  if (t == 0) rmax = red[0];
  __syncthreads();
  float ea = __expf(a - rmax), eb = __expf(b - rmax);
  red[t] = ea + eb; __syncthreads();
  for (int s = 128; s > 0; s >>= 1) { if (t < s) red[t] += red[t + s]; __syncthreads(); }
  if (t == 0) rsum = red[0];
  __syncthreads();
  float inv = 1.0f / rsum;
  if (masked) {
    p[t]       = (a != 0.0f) ? ea * inv : 0.0f;
    p[t + 256] = (b != 0.0f) ? eb * inv : 0.0f;
  } else {
    p[t] = ea * inv;
    p[t + 256] = eb * inv;
  }
}

// ---------------------------------------------------------------------------
// decoder token + time-of-day embedding
// ---------------------------------------------------------------------------
__global__ __launch_bounds__(256) void dec_embed_kernel(
    const float* __restrict__ demb, const float* __restrict__ temb,
    const int* __restrict__ dtime, float* __restrict__ D) {
  long idx = blockIdx.x;             // b*512 + n
  int b = (int)(idx >> 9), n = (int)(idx & 511);
  int e = threadIdx.x;
  D[idx * 256 + e] = demb[n * 256 + e] + temb[(long)dtime[b] * 256 + e];
}

// ---------------------------------------------------------------------------
static inline G G0() {
  G g; memset(&g, 0, sizeof(g));
  g.alpha = 1.0f; g.innerN = 1; g.batches = 1; g.act = 0;
  return g;
}
static inline void rungemm(hipStream_t s, const G& g) {
  dim3 grid(g.N / 64, g.M / 64, g.batches);
  gemm_wmma_kernel<<<grid, dim3(128), 0, s>>>(g);
}

extern "C" void kernel_launch(void* const* d_in, const int* in_sizes, int n_in,
                              void* d_out, int out_size, void* d_ws, size_t ws_size,
                              hipStream_t stream) {
  // inputs in setup_inputs() dict order
  const float* enc_x    = (const float*)d_in[0];
  const int*   dec_time = (const int*)  d_in[1];
  const float* trans    = (const float*)d_in[2];
  const float* embed_o_w = (const float*)d_in[3];
  const float* embed_o_b = (const float*)d_in[4];
  const float* embed_d_w = (const float*)d_in[5];
  const float* embed_d_b = (const float*)d_in[6];
  const float* enc_n1_g = (const float*)d_in[7];
  const float* enc_n1_b = (const float*)d_in[8];
  const float* enc_qkv_w = (const float*)d_in[9];
  const float* enc_proj_w = (const float*)d_in[10];
  const float* enc_proj_b = (const float*)d_in[11];
  const float* enc_n2_g = (const float*)d_in[12];
  const float* enc_n2_b = (const float*)d_in[13];
  const float* enc_fc1_w = (const float*)d_in[14];
  const float* enc_fc1_b = (const float*)d_in[15];
  const float* enc_fc2_w = (const float*)d_in[16];
  const float* enc_fc2_b = (const float*)d_in[17];
  const float* dec_n1_g = (const float*)d_in[18];
  const float* dec_n1_b = (const float*)d_in[19];
  const float* dec_q_w = (const float*)d_in[20];
  const float* dec_k_w = (const float*)d_in[21];
  const float* dec_v_w = (const float*)d_in[22];
  const float* dec_proj_w = (const float*)d_in[23];
  const float* dec_proj_b = (const float*)d_in[24];
  const float* dec_n2_g = (const float*)d_in[25];
  const float* dec_n2_b = (const float*)d_in[26];
  const float* dec_fc1_w = (const float*)d_in[27];
  const float* dec_fc1_b = (const float*)d_in[28];
  const float* dec_fc2_w = (const float*)d_in[29];
  const float* dec_fc2_b = (const float*)d_in[30];
  const float* dec_embedding  = (const float*)d_in[31];
  const float* time_embedding = (const float*)d_in[32];
  const float* ph_fc1_w = (const float*)d_in[33];
  const float* ph_fc1_b = (const float*)d_in[34];
  const float* ph_fc2_w = (const float*)d_in[35];
  const float* ph_fc2_b = (const float*)d_in[36];
  (void)in_sizes; (void)n_in; (void)out_size; (void)ws_size;

  // workspace layout (floats); total ~33.9M floats (~135 MB)
  float* ws  = (float*)d_ws;
  float* SO  = ws;                       // [16,512,12]
  float* SD  = ws + 98304;               // [16,512,12]
  float* X   = ws + 262144;              // [8192,256] stream
  float* T1  = X  + 2097152;             // [8192,256]
  float* T2  = T1 + 2097152;             // [8192,256]
  float* AO  = T2 + 2097152;             // [8192,256] attn out
  float* DEC = AO + 2097152;             // [8192,256] dec embed
  float* QKV = DEC + 2097152;            // [8192,768]
  float* S   = QKV + 6291456;            // [64,512,512] scores

  const long BATA = 512L * 768;          // per-b stride in QKV
  const long SB   = 262144;              // 512*512

  // ---- STAR embed ----
  colsum_kernel<<<dim3(192),  dim3(512), 0, stream>>>(enc_x, SO);
  rowsum_kernel<<<dim3(12288), dim3(256), 0, stream>>>(enc_x, SD);
  embed_kernel<<<dim3(8192), dim3(256), 0, stream>>>(
      SO, SD, embed_o_w, embed_o_b, embed_d_w, embed_d_b, X);

  // ---- Encoder block ----
  layernorm_kernel<<<dim3(8192), dim3(256), 0, stream>>>(X, T1, enc_n1_g, enc_n1_b);
  { G g = G0(); g.A = T1; g.sAm = 256; g.B = enc_qkv_w; g.sBk = 768; g.sBn = 1;
    g.C = QKV; g.sCm = 768; g.M = 8192; g.N = 768; g.K = 256; rungemm(stream, g); }
  // scores = scale * Q @ K^T   (per b,h)
  { G g = G0(); g.innerN = 4; g.batches = 64;
    g.A = QKV;       g.Ao = BATA; g.Ai = 64; g.sAm = 768;
    g.B = QKV + 256; g.Bo = BATA; g.Bi = 64; g.sBk = 1; g.sBn = 768; // K^T
    g.C = S; g.Co = 4 * SB; g.Ci = SB; g.sCm = 512;
    g.alpha = 0.125f; g.M = 512; g.N = 512; g.K = 64; rungemm(stream, g); }
  softmax_kernel<<<dim3(32768), dim3(256), 0, stream>>>(S, 0);
  // AO = P @ V  (merged heads)
  { G g = G0(); g.innerN = 4; g.batches = 64;
    g.A = S;         g.Ao = 4 * SB; g.Ai = SB; g.sAm = 512;
    g.B = QKV + 512; g.Bo = BATA;   g.Bi = 64; g.sBk = 768; g.sBn = 1;
    g.C = AO; g.Co = 512L * 256; g.Ci = 64; g.sCm = 256;
    g.M = 512; g.N = 64; g.K = 512; rungemm(stream, g); }
  // X = X + AO @ proj + b
  { G g = G0(); g.A = AO; g.sAm = 256; g.B = enc_proj_w; g.sBk = 256; g.sBn = 1;
    g.C = X; g.sCm = 256; g.bias = enc_proj_b; g.R = X; g.sRm = 256;
    g.M = 8192; g.N = 256; g.K = 256; rungemm(stream, g); }
  layernorm_kernel<<<dim3(8192), dim3(256), 0, stream>>>(X, T1, enc_n2_g, enc_n2_b);
  { G g = G0(); g.A = T1; g.sAm = 256; g.B = enc_fc1_w; g.sBk = 256; g.sBn = 1;
    g.C = T2; g.sCm = 256; g.bias = enc_fc1_b; g.act = 1;
    g.M = 8192; g.N = 256; g.K = 256; rungemm(stream, g); }
  { G g = G0(); g.A = T2; g.sAm = 256; g.B = enc_fc2_w; g.sBk = 256; g.sBn = 1;
    g.C = X; g.sCm = 256; g.bias = enc_fc2_b; g.R = X; g.sRm = 256;
    g.M = 8192; g.N = 256; g.K = 256; rungemm(stream, g); }   // X = enc_outputs

  // ---- Decoder ----
  dec_embed_kernel<<<dim3(8192), dim3(256), 0, stream>>>(
      dec_embedding, time_embedding, dec_time, DEC);
  layernorm_kernel<<<dim3(8192), dim3(256), 0, stream>>>(DEC, T1, dec_n1_g, dec_n1_b); // hd
  layernorm_kernel<<<dim3(8192), dim3(256), 0, stream>>>(X,   T2, dec_n1_g, dec_n1_b); // he
  { G g = G0(); g.A = T1; g.sAm = 256; g.B = dec_q_w; g.sBk = 256; g.sBn = 1;
    g.C = QKV;       g.sCm = 768; g.M = 8192; g.N = 256; g.K = 256; rungemm(stream, g); }
  { G g = G0(); g.A = T2; g.sAm = 256; g.B = dec_k_w; g.sBk = 256; g.sBn = 1;
    g.C = QKV + 256; g.sCm = 768; g.M = 8192; g.N = 256; g.K = 256; rungemm(stream, g); }
  { G g = G0(); g.A = T2; g.sAm = 256; g.B = dec_v_w; g.sBk = 256; g.sBn = 1;
    g.C = QKV + 512; g.sCm = 768; g.M = 8192; g.N = 256; g.K = 256; rungemm(stream, g); }
  // masked scores = (scale * Q @ K^T) * trans[b]
  { G g = G0(); g.innerN = 4; g.batches = 64;
    g.A = QKV;       g.Ao = BATA; g.Ai = 64; g.sAm = 768;
    g.B = QKV + 256; g.Bo = BATA; g.Bi = 64; g.sBk = 1; g.sBn = 768;
    g.C = S; g.Co = 4 * SB; g.Ci = SB; g.sCm = 512;
    g.Mk = trans; g.Mko = SB; g.sMm = 512;
    g.alpha = 0.125f; g.M = 512; g.N = 512; g.K = 64; rungemm(stream, g); }
  softmax_kernel<<<dim3(32768), dim3(256), 0, stream>>>(S, 1);
  { G g = G0(); g.innerN = 4; g.batches = 64;
    g.A = S;         g.Ao = 4 * SB; g.Ai = SB; g.sAm = 512;
    g.B = QKV + 512; g.Bo = BATA;   g.Bi = 64; g.sBk = 768; g.sBn = 1;
    g.C = AO; g.Co = 512L * 256; g.Ci = 64; g.sCm = 256;
    g.M = 512; g.N = 64; g.K = 512; rungemm(stream, g); }
  // d = AO @ dec_proj + b + DEC   -> X
  { G g = G0(); g.A = AO; g.sAm = 256; g.B = dec_proj_w; g.sBk = 256; g.sBn = 1;
    g.C = X; g.sCm = 256; g.bias = dec_proj_b; g.R = DEC; g.sRm = 256;
    g.M = 8192; g.N = 256; g.K = 256; rungemm(stream, g); }
  layernorm_kernel<<<dim3(8192), dim3(256), 0, stream>>>(X, T1, dec_n2_g, dec_n2_b);
  { G g = G0(); g.A = T1; g.sAm = 256; g.B = dec_fc1_w; g.sBk = 256; g.sBn = 1;
    g.C = T2; g.sCm = 256; g.bias = dec_fc1_b; g.act = 1;
    g.M = 8192; g.N = 256; g.K = 256; rungemm(stream, g); }
  { G g = G0(); g.A = T2; g.sAm = 256; g.B = dec_fc2_w; g.sBk = 256; g.sBn = 1;
    g.C = X; g.sCm = 256; g.bias = dec_fc2_b;                       // no residual
    g.M = 8192; g.N = 256; g.K = 256; rungemm(stream, g); }

  // ---- PHEAD ----
  { G g = G0(); g.A = X; g.sAm = 256; g.B = ph_fc1_w; g.sBk = 256; g.sBn = 1;
    g.C = T1; g.sCm = 256; g.bias = ph_fc1_b; g.act = 2;            // ReLU
    g.M = 8192; g.N = 256; g.K = 256; rungemm(stream, g); }
  { G g = G0(); g.A = T1; g.sAm = 256; g.B = ph_fc2_w; g.sBk = 512; g.sBn = 1;
    g.C = (float*)d_out; g.sCm = 512; g.bias = ph_fc2_b;
    g.M = 8192; g.N = 512; g.K = 256; rungemm(stream, g); }         // [16,1,512,512]
}
